// SchNet_4750233830155
// MI455X (gfx1250) — matrix operand order repaired and checked by reading, not compiled, not run on hardware
//
#include <hip/hip_runtime.h>
#include <hip/hip_fp16.h>

typedef __attribute__((ext_vector_type(16))) _Float16 v16h;
typedef __attribute__((ext_vector_type(8)))  float    v8f;

#define CUTOFF_F 5.0f

__device__ __forceinline__ float sspf(float x) {
  // softplus(x) - log(2), numerically stable; sspf(0) == 0 exactly
  return fmaxf(x, 0.0f) + log1pf(__expf(-fabsf(x))) - 0.6931471805599453f;
}

// ---------------------------------------------------------------- trivial ---
__global__ void zero_kernel(float* __restrict__ p, int n) {
  int i = blockIdx.x * 256 + threadIdx.x;
  if (i < n) p[i] = 0.0f;
}

__global__ void embed_kernel(const int* __restrict__ r, const float* __restrict__ embed,
                             float* __restrict__ h, int nAtoms) {
  int idx = blockIdx.x * 256 + threadIdx.x;
  if (idx < nAtoms * 128) {
    int atom = idx >> 7, c = idx & 127;
    h[idx] = embed[r[atom] * 128 + c];
  }
}

__global__ void dist_kernel(const float* __restrict__ xyz, const int* __restrict__ edges,
                            float* __restrict__ d, int nEdges) {
  int e = blockIdx.x * 256 + threadIdx.x;
  if (e < nEdges) {
    int i = edges[2 * e], j = edges[2 * e + 1];
    float dx = xyz[3 * i + 0] - xyz[3 * j + 0];
    float dy = xyz[3 * i + 1] - xyz[3 * j + 1];
    float dz = xyz[3 * i + 2] - xyz[3 * j + 2];
    d[e] = sqrtf(dx * dx + dy * dy + dz * dz);
  }
}

// ------------------------------------------------- generic WMMA GEMM -------
// out[M,N] = act(A[M,K] @ W[K,N] + b[N]) (+ res[M,N] if RES)
// 4 waves / 128 threads per block; each wave owns a 16-row strip.
template <int K, int N, bool ACT, bool RES>
__global__ void __launch_bounds__(128)
gemm_rowblock(const float* __restrict__ A, const float* __restrict__ W,
              const float* __restrict__ b, const float* __restrict__ res,
              float* __restrict__ out, int M) {
  __shared__ _Float16 sA[64][K];
  __shared__ _Float16 sW[K][N];
  const int tid = threadIdx.x;
  const int wave = tid >> 5, lane = tid & 31;
  const int row0 = blockIdx.x * 64;

  for (int idx = tid; idx < K * N; idx += 128)
    sW[idx / N][idx % N] = (_Float16)W[idx];
  for (int idx = tid; idx < 64 * K; idx += 128) {
    int rr = idx / K, cc = idx % K;
    int gr = row0 + rr;
    sA[rr][cc] = (_Float16)(gr < M ? A[(size_t)gr * K + cc] : 0.0f);
  }
  __syncthreads();

  const int mrow = wave * 16 + (lane & 15);
  const int hsel = lane >> 4;  // which half of the wave
  const int col  = lane & 15;

  // A fragments (16x32 f16 per K-tile), contiguous k = 16*hsel + s order
  v16h afrag[K / 32];
#pragma unroll
  for (int kt = 0; kt < K / 32; ++kt) {
#pragma unroll
    for (int v = 0; v < 8; ++v) {
      int k = kt * 32 + hsel * 16 + 2 * v;
      afrag[kt][2 * v]     = sA[mrow][k];
      afrag[kt][2 * v + 1] = sA[mrow][k + 1];
    }
  }

#pragma unroll
  for (int nt = 0; nt < N / 16; ++nt) {
    v8f acc;
    float bias = b[nt * 16 + col];
#pragma unroll
    for (int v = 0; v < 8; ++v) acc[v] = bias;
#pragma unroll
    for (int kt = 0; kt < K / 32; ++kt) {
      v16h bfrag;
      int kb = kt * 32 + hsel * 16;
#pragma unroll
      for (int v = 0; v < 8; ++v) {
        bfrag[2 * v]     = sW[kb + 2 * v][nt * 16 + col];
        bfrag[2 * v + 1] = sW[kb + 2 * v + 1][nt * 16 + col];
      }
      acc = __builtin_amdgcn_wmma_f32_16x16x32_f16(false, afrag[kt], false, bfrag,
                                                   (short)0, acc, false, false);
    }
#pragma unroll
    for (int v = 0; v < 8; ++v) {
      int m = row0 + wave * 16 + v + 8 * hsel;
      int n = nt * 16 + col;
      if (m < M) {
        float x = acc[v];
        if (ACT) x = sspf(x);
        if (RES) x += res[(size_t)m * N + n];
        out[(size_t)m * N + n] = x;
      }
    }
  }
}

// ------------------------------------------------ fused edge-MLP kernel ----
// Per edge: g(50 gauss) -> ssp(g@We1+be1) -> @We2+be2 = ef(128)
// then atomicAdd(agg[i], ef*nodef[j]) and atomicAdd(agg[j], ef*nodef[i]).
// 8 waves / 256 threads per block; 16 edges per wave; K/N padded 50->64 (zeros).
__global__ void __launch_bounds__(256)
edge_kernel(const float* __restrict__ d, const int* __restrict__ edges,
            const float* __restrict__ nodef,
            const float* __restrict__ We1, const float* __restrict__ be1,
            const float* __restrict__ We2, const float* __restrict__ be2,
            float* __restrict__ agg, int nEdges) {
  __shared__ _Float16 sW1[64][64];       // We1 zero-padded (8 KB)
  __shared__ _Float16 sW2[64][128];      // We2 zero-padded (16 KB)
  __shared__ _Float16 sT[8][16][64];     // per-wave ssp(g@We1+be1) restage (16 KB)
  __shared__ float    sBe1[64];

  const int tid = threadIdx.x, wave = tid >> 5, lane = tid & 31;
  const int hsel = lane >> 4, mrow = lane & 15, col = lane & 15;

  for (int idx = tid; idx < 64 * 64; idx += 256) {
    int k = idx >> 6, n = idx & 63;
    sW1[k][n] = (_Float16)((k < 50 && n < 50) ? We1[k * 50 + n] : 0.0f);
  }
  for (int idx = tid; idx < 64 * 128; idx += 256) {
    int k = idx >> 7, n = idx & 127;
    sW2[k][n] = (_Float16)((k < 50) ? We2[k * 128 + n] : 0.0f);
  }
  if (tid < 64) sBe1[tid] = (tid < 50) ? be1[tid] : 0.0f;
  __syncthreads();

  const int ebase = blockIdx.x * 128 + wave * 16;

  // Gaussian basis straight into the A-fragment registers (no LDS pass).
  int eC = ebase + mrow;
  if (eC >= nEdges) eC = nEdges - 1;
  const float dm = d[eC];
  const float kw = CUTOFF_F / 49.0f;           // linspace step
  const float coeff = -0.5f / (kw * kw);

  v16h ga[2];
#pragma unroll
  for (int kt = 0; kt < 2; ++kt) {
#pragma unroll
    for (int v = 0; v < 8; ++v) {
      int k0 = kt * 32 + hsel * 16 + 2 * v;
#pragma unroll
      for (int u = 0; u < 2; ++u) {
        int k = k0 + u;
        float val = 0.0f;
        if (k < 50) {
          float t = dm - kw * (float)k;
          val = __expf(coeff * t * t);
        }
        ga[kt][2 * v + u] = (_Float16)val;
      }
    }
  }

  // GEMM1: t(16x64) = ssp(g @ We1 + be1)  -> sT[wave]
#pragma unroll
  for (int nt = 0; nt < 4; ++nt) {
    v8f acc;
    float bias = sBe1[nt * 16 + col];
#pragma unroll
    for (int v = 0; v < 8; ++v) acc[v] = bias;
#pragma unroll
    for (int kt = 0; kt < 2; ++kt) {
      v16h bfrag;
      int kb = kt * 32 + hsel * 16;
#pragma unroll
      for (int v = 0; v < 8; ++v) {
        bfrag[2 * v]     = sW1[kb + 2 * v][nt * 16 + col];
        bfrag[2 * v + 1] = sW1[kb + 2 * v + 1][nt * 16 + col];
      }
      acc = __builtin_amdgcn_wmma_f32_16x16x32_f16(false, ga[kt], false, bfrag,
                                                   (short)0, acc, false, false);
    }
#pragma unroll
    for (int v = 0; v < 8; ++v)
      sT[wave][v + 8 * hsel][nt * 16 + col] = (_Float16)sspf(acc[v]);
  }
  __syncthreads();

  // GEMM2: ef(16x128) = t @ We2 + be2 (kept in C fragments)
  v16h ta[2];
#pragma unroll
  for (int kt = 0; kt < 2; ++kt) {
#pragma unroll
    for (int v = 0; v < 8; ++v) {
      int k = kt * 32 + hsel * 16 + 2 * v;
      ta[kt][2 * v]     = sT[wave][mrow][k];
      ta[kt][2 * v + 1] = sT[wave][mrow][k + 1];
    }
  }

  v8f ef[8];
#pragma unroll
  for (int nt = 0; nt < 8; ++nt) {
    v8f acc;
    float bias = be2[nt * 16 + col];
#pragma unroll
    for (int v = 0; v < 8; ++v) acc[v] = bias;
#pragma unroll
    for (int kt = 0; kt < 2; ++kt) {
      v16h bfrag;
      int kb = kt * 32 + hsel * 16;
#pragma unroll
      for (int v = 0; v < 8; ++v) {
        bfrag[2 * v]     = sW2[kb + 2 * v][nt * 16 + col];
        bfrag[2 * v + 1] = sW2[kb + 2 * v + 1][nt * 16 + col];
      }
      acc = __builtin_amdgcn_wmma_f32_16x16x32_f16(false, ta[kt], false, bfrag,
                                                   (short)0, acc, false, false);
    }
    ef[nt] = acc;
  }

  // Scatter: for C-fragment row v, 16 consecutive lanes hit 16 consecutive
  // columns of one agg row -> 64B-coalesced atomic segments (L2-resident).
#pragma unroll
  for (int v = 0; v < 8; ++v) {
    int e = ebase + v + 8 * hsel;
    if (e < nEdges) {
      int ii = edges[2 * e], jj = edges[2 * e + 1];
      const float* nfi = nodef + (size_t)ii * 128;
      const float* nfj = nodef + (size_t)jj * 128;
      float* gi = agg + (size_t)ii * 128;
      float* gj = agg + (size_t)jj * 128;
#pragma unroll
      for (int nt = 0; nt < 8; ++nt) {
        int n = nt * 16 + col;
        float f = ef[nt][v];
        atomicAdd(gi + n, f * nfj[n]);
        atomicAdd(gj + n, f * nfi[n]);
      }
    }
  }
}

// ------------------------------------------------ molecule readout ---------
// out[mol] = sum over atoms of (t2[atom,0:64] . W2 + b2); one block per molecule
__global__ void __launch_bounds__(256)
mol_energy_kernel(const float* __restrict__ t2, const float* __restrict__ W2,
                  const float* __restrict__ b2, float* __restrict__ out,
                  int molSize, int nAtoms) {
  __shared__ float sw[64];
  __shared__ float red[256];
  if (threadIdx.x < 64) sw[threadIdx.x] = W2[threadIdx.x];
  __syncthreads();
  int mol = blockIdx.x;
  int aEnd = (mol + 1) * molSize;
  if (aEnd > nAtoms) aEnd = nAtoms;
  float sum = 0.0f;
  for (int a = mol * molSize + threadIdx.x; a < aEnd; a += 256) {
    float dot = b2[0];
#pragma unroll 8
    for (int c = 0; c < 64; ++c) dot += t2[(size_t)a * 64 + c] * sw[c];
    sum += dot;
  }
  red[threadIdx.x] = sum;
  __syncthreads();
  for (int s = 128; s > 0; s >>= 1) {
    if (threadIdx.x < s) red[threadIdx.x] += red[threadIdx.x + s];
    __syncthreads();
  }
  if (threadIdx.x == 0) out[mol] = red[0];
}

// ---------------------------------------------------------------------------
extern "C" void kernel_launch(void* const* d_in, const int* in_sizes, int n_in,
                              void* d_out, int out_size, void* d_ws, size_t ws_size,
                              hipStream_t stream) {
  const int*   r     = (const int*)d_in[0];
  const float* xyz   = (const float*)d_in[1];
  const int*   edges = (const int*)d_in[2];
  const float* embed = (const float*)d_in[4];
  const float* Wn    = (const float*)d_in[5];
  const float* bn    = (const float*)d_in[6];
  const float* We1   = (const float*)d_in[7];
  const float* be1   = (const float*)d_in[8];
  const float* We2   = (const float*)d_in[9];
  const float* be2   = (const float*)d_in[10];
  const float* Wu1   = (const float*)d_in[11];
  const float* bu1   = (const float*)d_in[12];
  const float* Wu2   = (const float*)d_in[13];
  const float* bu2   = (const float*)d_in[14];
  const float* W1    = (const float*)d_in[15];
  const float* b1    = (const float*)d_in[16];
  const float* W2    = (const float*)d_in[17];
  const float* b2    = (const float*)d_in[18];
  float* out = (float*)d_out;

  const int nAtoms  = in_sizes[0];       // 50000
  const int nEdges  = in_sizes[2] / 2;   // 500000
  const int molSize = 1000;              // reference MOL_SIZE

  char* ws = (char*)d_ws;
  size_t off = 0;
  auto alloc = [&](size_t bytes) -> void* {
    void* p = ws + off;
    off += (bytes + 255) & ~(size_t)255;
    return p;
  };
  float* h     = (float*)alloc((size_t)nAtoms * 128 * sizeof(float));
  float* nodef = (float*)alloc((size_t)nAtoms * 128 * sizeof(float)); // reused as temp
  float* agg   = (float*)alloc((size_t)nAtoms * 128 * sizeof(float));
  float* dbuf  = (float*)alloc((size_t)nEdges * sizeof(float));

  embed_kernel<<<(nAtoms * 128 + 255) / 256, 256, 0, stream>>>(r, embed, h, nAtoms);
  dist_kernel<<<(nEdges + 255) / 256, 256, 0, stream>>>(xyz, edges, dbuf, nEdges);

  const int gemmBlocks = (nAtoms + 63) / 64;
  const int edgeBlocks = (nEdges + 127) / 128;

  for (int l = 0; l < 3; ++l) {
    zero_kernel<<<(nAtoms * 128 + 255) / 256, 256, 0, stream>>>(agg, nAtoms * 128);
    // nodef = h @ Wn[l] + bn[l]
    gemm_rowblock<128, 128, false, false><<<gemmBlocks, 128, 0, stream>>>(
        h, Wn + (size_t)l * 128 * 128, bn + (size_t)l * 128, nullptr, nodef, nAtoms);
    // fused edge MLP + bidirectional aggregation
    edge_kernel<<<edgeBlocks, 256, 0, stream>>>(
        dbuf, edges, nodef, We1 + (size_t)l * 50 * 50, be1 + (size_t)l * 50,
        We2 + (size_t)l * 50 * 128, be2 + (size_t)l * 128, agg, nEdges);
    // nodef (as temp) = ssp(agg @ Wu1[l] + bu1[l])
    gemm_rowblock<128, 128, true, false><<<gemmBlocks, 128, 0, stream>>>(
        agg, Wu1 + (size_t)l * 128 * 128, bu1 + (size_t)l * 128, nullptr, nodef, nAtoms);
    // h = h + (temp @ Wu2[l] + bu2[l])
    gemm_rowblock<128, 128, false, true><<<gemmBlocks, 128, 0, stream>>>(
        nodef, Wu2 + (size_t)l * 128 * 128, bu2 + (size_t)l * 128, h, h, nAtoms);
  }

  // nodef (as temp, stride 64) = ssp(h @ W1 + b1)
  gemm_rowblock<128, 64, true, false><<<gemmBlocks, 128, 0, stream>>>(
      h, W1, b1, nullptr, nodef, nAtoms);

  mol_energy_kernel<<<nAtoms / molSize, 256, 0, stream>>>(nodef, W2, b2, out,
                                                          molSize, nAtoms);
}
